// GCN_S_15977278341730
// MI455X (gfx1250) — matrix operand order repaired and checked by reading, not compiled, not run on hardware
//
#include <hip/hip_runtime.h>
#include <hip/hip_bf16.h>
#include <math.h>

#define EMB 64

typedef __attribute__((ext_vector_type(16))) __bf16 v16bf;
typedef __attribute__((ext_vector_type(8)))  float  v8f;

// ---------------------------------------------------------------------------
// Zero a float buffer (ws is poisoned 0xAA; SpMM scatter-adds need zeros).
// ---------------------------------------------------------------------------
__global__ void zero_kernel(float* __restrict__ p, int n) {
    int i = blockIdx.x * blockDim.x + threadIdx.x;
    if (i < n) p[i] = 0.0f;
}

// ---------------------------------------------------------------------------
// COO SpMM: out[r] += vals[e] * x[cols[e]]   (out pre-zeroed)
// One wave (32 lanes) per edge; each lane handles 2 of 64 embedding floats.
// x row (256B) and out row are coalesced; atomics land in L2 (table is L2-resident).
// ---------------------------------------------------------------------------
__global__ void spmm_kernel(const int* __restrict__ rows, const int* __restrict__ cols,
                            const float* __restrict__ vals, const float* __restrict__ x,
                            float* __restrict__ out, int nedges) {
    long long t = (long long)blockIdx.x * blockDim.x + threadIdx.x;
    int e    = (int)(t >> 5);
    int lane = (int)(t & 31);
    if (e >= nedges) return;
    int r = rows[e];
    int c = cols[e];
    float v = vals[e];
    const float* xr = x + (long long)c * EMB;
    float* orow     = out + (long long)r * EMB;
    atomicAdd(&orow[lane],      v * xr[lane]);
    atomicAdd(&orow[lane + 32], v * xr[lane + 32]);
}

// ---------------------------------------------------------------------------
// Y = l2norm_rows( leaky_relu( X @ W^T + b ) )
// Block = 128 threads (4 waves) = one 16-row tile.
// Wave w computes columns [16w, 16w+16) as a 16x16 f32 tile:
//   two v_wmma_f32_16x16x32_bf16 over K=64.
// Fragment layouts used here (wave32; A and B share contiguous K order):
//   A (16x32 bf16): lane = M (mod 16); half = lane>>4.
//     vector elems 0..7  -> K = k0 + 16*half + j
//     vector elems 8..15 -> K = k0 + 16*half + 8 + j
//   B (32x16 bf16): lane = N (mod 16); elems j -> K = k0 + 16*half + j.
//     Since B[k][n] = W[n][k], a lane's 16 halves are 16 contiguous floats
//     of row (16w + n) of W.
//   C/D (16x16 f32): elem v -> M = v + 8*half, N = 16w + (lane&15).
// nrows must be a multiple of 16 (100000 and 50000 both are) so EXEC is all 1s.
// ---------------------------------------------------------------------------
__global__ __launch_bounds__(128)
void dense_lrelu_l2norm_kernel(const float* __restrict__ X, const float* __restrict__ W,
                               const float* __restrict__ bias, float* __restrict__ out) {
    __shared__ float tile[16][EMB];
    __shared__ float invnorm[16];

    const int tid  = threadIdx.x;
    const int wave = tid >> 5;        // 0..3 -> column tile
    const int lane = tid & 31;
    const int half = lane >> 4;       // 0|1
    const int lm   = lane & 15;
    const int rowBase = blockIdx.x * 16;

    v8f acc = {0.f, 0.f, 0.f, 0.f, 0.f, 0.f, 0.f, 0.f};

    const float* xrow = X + (long long)(rowBase + lm) * EMB;

#pragma unroll
    for (int kk = 0; kk < 2; ++kk) {
        const int k0 = kk * 32;
        v16bf a, b;
        // A fragment: row M = rowBase + lm
#pragma unroll
        for (int j = 0; j < 8; ++j) {
            a[j]     = (__bf16)xrow[k0 + 16 * half + j];
            a[j + 8] = (__bf16)xrow[k0 + 16 * half + 8 + j];
        }
        // B fragment: column N = 16*wave + lm; 16 contiguous K values of W row N
        const float* wr = W + (long long)(16 * wave + lm) * EMB + k0 + 16 * half;
#pragma unroll
        for (int j = 0; j < 16; ++j) b[j] = (__bf16)wr[j];

        acc = __builtin_amdgcn_wmma_f32_16x16x32_bf16(
            /*neg_a=*/false, a, /*neg_b=*/false, b,
            /*c_mod=*/(short)0, acc, /*reuse_a=*/false, /*reuse_b=*/false);
    }

    // bias + leaky_relu into LDS tile
    const int col  = 16 * wave + lm;
    const float bc = bias[col];
#pragma unroll
    for (int v = 0; v < 8; ++v) {
        int m   = v + 8 * half;
        float y = acc[v] + bc;
        y = (y > 0.0f) ? y : 0.01f * y;
        tile[m][col] = y;
    }
    __syncthreads();

    // per-row L2 norm (16 rows, thread t<16 handles row t)
    if (tid < 16) {
        float s = 0.0f;
#pragma unroll
        for (int c = 0; c < EMB; ++c) { float t = tile[tid][c]; s += t * t; }
        float n = sqrtf(s);
        invnorm[tid] = 1.0f / fmaxf(n, 1e-12f);
    }
    __syncthreads();

    // normalized coalesced writeback: 1024 elems / 128 threads = 8 each
#pragma unroll
    for (int i = 0; i < 8; ++i) {
        int idx = tid + 128 * i;
        int m = idx >> 6, c = idx & 63;
        out[(long long)(rowBase + m) * EMB + c] = tile[m][c] * invnorm[m];
    }
}

// ---------------------------------------------------------------------------
// Host orchestration (graph-capture safe: only kernel launches on `stream`)
// ---------------------------------------------------------------------------
static inline void run_spmm(const int* rows, const int* cols, const float* vals,
                            const float* x, float* out, int n_nodes, int n_edges,
                            hipStream_t stream) {
    int nz = n_nodes * EMB;
    zero_kernel<<<(nz + 255) / 256, 256, 0, stream>>>(out, nz);
    long long threads = (long long)n_edges * 32;
    int blocks = (int)((threads + 255) / 256);
    spmm_kernel<<<blocks, 256, 0, stream>>>(rows, cols, vals, x, out, n_edges);
}

extern "C" void kernel_launch(void* const* d_in, const int* in_sizes, int n_in,
                              void* d_out, int out_size, void* d_ws, size_t ws_size,
                              hipStream_t stream) {
    const float* user_emb = (const float*)d_in[0];
    const float* item_emb = (const float*)d_in[1];
    const float* Wu0 = (const float*)d_in[2];
    const float* bu0 = (const float*)d_in[3];
    const float* Wu1 = (const float*)d_in[4];
    const float* bu1 = (const float*)d_in[5];
    const float* Wi0 = (const float*)d_in[6];
    const float* bi0 = (const float*)d_in[7];
    const float* Wi1 = (const float*)d_in[8];
    const float* bi1 = (const float*)d_in[9];
    const int*   u_rows = (const int*)d_in[10];
    const int*   u_cols = (const int*)d_in[11];
    const float* u_vals = (const float*)d_in[12];
    const int*   i_rows = (const int*)d_in[13];
    const int*   i_cols = (const int*)d_in[14];
    const float* i_vals = (const float*)d_in[15];

    const int NU = in_sizes[0] / EMB;   // 100000
    const int NI = in_sizes[1] / EMB;   // 50000
    const int EU = in_sizes[10];        // 1600000
    const int EI = in_sizes[13];        // 800000

    // workspace layout (floats): su | cu | si | ci
    float* ws = (float*)d_ws;
    float* su = ws;
    float* cu = su + (size_t)NU * EMB;
    float* si = cu + (size_t)NU * EMB;
    float* ci = si + (size_t)NI * EMB;

    float* out_u = (float*)d_out;
    float* out_i = out_u + (size_t)NU * EMB;

    const int gu = NU / 16;  // 6250 blocks (exact)
    const int gi = NI / 16;  // 3125 blocks (exact)

    // ---- layer 0 ----
    run_spmm(u_rows, u_cols, u_vals, user_emb, su, NU, EU, stream);
    dense_lrelu_l2norm_kernel<<<gu, 128, 0, stream>>>(su, Wu0, bu0, cu);

    run_spmm(i_rows, i_cols, i_vals, item_emb, si, NI, EI, stream);
    dense_lrelu_l2norm_kernel<<<gi, 128, 0, stream>>>(si, Wi0, bi0, ci);

    // ---- layer 1 ----
    run_spmm(u_rows, u_cols, u_vals, cu, su, NU, EU, stream);
    dense_lrelu_l2norm_kernel<<<gu, 128, 0, stream>>>(su, Wu1, bu1, out_u);

    run_spmm(i_rows, i_cols, i_vals, ci, si, NI, EI, stream);
    dense_lrelu_l2norm_kernel<<<gi, 128, 0, stream>>>(si, Wi1, bi1, out_i);
}
